// HierarchicalBiLSTM_CRF_32976758898924
// MI455X (gfx1250) — matrix operand order repaired and checked by reading, not compiled, not run on hardware
//
#include <hip/hip_runtime.h>
#include <hip/hip_bf16.h>

// ---------------------------------------------------------------------------
// Hierarchical BiLSTM-CRF for MI455X (gfx1250, wave32, WMMA).
//
// Algebraic reductions (exact w.r.t. the reference):
//  * token forward:  last = h[clen-1] == final h of the masked 64-step scan.
//  * token backward: last = hb[clen-1] == ONE LSTM step on x[clen-1] from 0
//    state (masked reverse scan keeps h=0 until t=clen-1), so no recurrence.
//  * chunk level: full 32-step masked BiLSTM, outputs zeroed at pads.
//
// GEMM strategy: per block, 16 sequences (M=16). gates[16,1024] =
// [x_t | h](16xK) @ Wcat^T via v_wmma_f32_16x16x32_bf16. Wave w owns gate
// tiles {w,w+8,...,w+56} so each lane holds i,f,g,o of the same hidden unit
// -> LSTM cell fully in registers, c-state in VGPRs, h written to LDS bf16.
// Weights pre-swizzled to the B-operand lane layout (one 32B load / frag).
// Workspace requirement: ~26.5 MB.
// ---------------------------------------------------------------------------

typedef __attribute__((ext_vector_type(16))) __bf16        v16bf;
typedef __attribute__((ext_vector_type(8)))  float         v8f;
typedef __attribute__((ext_vector_type(4)))  unsigned int  u32x4;
typedef __attribute__((ext_vector_type(8)))  unsigned int  u32x8;

#define V_    30000
#define E_    300
#define HT_   256
#define B_    32
#define LC_   32
#define T_    64
#define NSEQ  (B_*LC_)      // 1024
#define NTAG_ 9
#define NB    64            // 1024 gates / 16
#define KB_TOKF 18          // K = 576 = 300(emb)+256(h)+20(pad)
#define KB_TOKB 10          // K = 320 = 300(emb)+20(pad)
#define KB_CHK  24          // K = 768 = 512(in)+256(h)
#define ASTR_F 584          // 576+8 pad (bank-conflict-free, 16B aligned rows)
#define ASTR_B 328          // 320+8
#define ASTR_C 776          // 768+8

#define DEVFN static __device__ __forceinline__

DEVFN unsigned short f2bf(float f) {
  unsigned u = __float_as_uint(f);
  unsigned r = ((u >> 16) & 1u) + 0x7FFFu;   // round-to-nearest-even
  return (unsigned short)((u + r) >> 16);
}
DEVFN float sigm(float x) { return 1.0f / (1.0f + __expf(-x)); }

DEVFN v16bf fr2(u32x4 a, u32x4 b) {
  u32x8 t = __builtin_shufflevector(a, b, 0, 1, 2, 3, 4, 5, 6, 7);
  return __builtin_bit_cast(v16bf, t);
}
DEVFN v16bf fr1(u32x8 a) { return __builtin_bit_cast(v16bf, a); }
DEVFN v8f wmma_bf16(v16bf a, v16bf b, v8f c) {
  return __builtin_amdgcn_wmma_f32_16x16x32_bf16(false, a, false, b,
                                                 (short)0, c, false, false);
}
DEVFN int clampT(int c) { return c < 1 ? 1 : (c > T_ ? T_ : c); }

// ---------------------------------------------------------------------------
// Prep kernels
// ---------------------------------------------------------------------------
__global__ void k_f32_to_bf16(const float* __restrict__ src,
                              unsigned short* __restrict__ dst, int n) {
  for (int i = blockIdx.x * blockDim.x + threadIdx.x; i < n;
       i += gridDim.x * blockDim.x)
    dst[i] = f2bf(src[i]);
}

// Swizzle [Wih | Whh] (row-major [1024][K]) into per-tile B-operand layout:
// dst[((nb*KBt + kb)*32 + lane)*16 + e]; lane -> N=gate col, K = kb*32 + 8*half + (e<8 ? e : 8+e).
__global__ void k_swizzle_w(const float* __restrict__ Wih,
                            const float* __restrict__ Whh,
                            int Kih, int Khh, int KBt,
                            unsigned short* __restrict__ dst) {
  int total = NB * KBt * 512;
  for (int i = blockIdx.x * blockDim.x + threadIdx.x; i < total;
       i += gridDim.x * blockDim.x) {
    int e    = i & 15;
    int lane = (i >> 4) & 31;
    int tile = i >> 9;
    int kb = tile % KBt, nb = tile / KBt;
    int n = lane & 15, half = lane >> 4;
    int K = kb * 32 + 8 * half + (e < 8 ? e : 8 + e);  // B-matrix 32x16 bf16 lane layout (table order)
    int g = nb * 16 + n;
    float v = 0.f;
    if (K < Kih)            v = Wih[g * Kih + K];
    else if (K < Kih + Khh) v = Whh[g * Khh + (K - Kih)];
    dst[i] = f2bf(v);
  }
}

// ---------------------------------------------------------------------------
// Token-level forward LSTM: 64 masked steps, emit final h (== h[clen-1]).
// grid 64 x 256; block handles sequences [16*blk, 16*blk+16).
// ---------------------------------------------------------------------------
__global__ __launch_bounds__(256) void k_token_fwd(
    const int* __restrict__ x, const int* __restrict__ xclen,
    const unsigned short* __restrict__ embbf,
    const unsigned short* __restrict__ Wt,   // swizzled [NB][KB_TOKF][512]
    const float* __restrict__ bias,          // [1024]
    unsigned short* __restrict__ chunks) {   // [NSEQ][512] bf16, cols 0..255
  __shared__ unsigned short As[16 * ASTR_F];
  const int tid = threadIdx.x;
  const int n0 = blockIdx.x * 16;
  const int w = tid >> 5, lane = tid & 31;
  const int n = lane & 15, half = lane >> 4;

  for (int i = tid; i < 16 * ASTR_F; i += 256) As[i] = 0;

  float bj[8];
#pragma unroll
  for (int j = 0; j < 8; ++j) bj[j] = bias[16 * (w + 8 * j) + n];
  int clen8[8];
#pragma unroll
  for (int r = 0; r < 8; ++r) clen8[r] = clampT(xclen[n0 + r + 8 * half]);
  float cst[2][8];
#pragma unroll
  for (int u = 0; u < 2; ++u)
#pragma unroll
    for (int r = 0; r < 8; ++r) cst[u][r] = 0.f;

  const int mrow = tid >> 4;   // embed-stage row (16 threads per row)
  const int l16 = tid & 15;

  for (int t = 0; t < T_; ++t) {
    {  // stage 1: embedding gather -> A[:,0:300) (bf16 pairs as u32)
      int tok = x[(n0 + mrow) * T_ + t];
      const unsigned int* ep = (const unsigned int*)(embbf + tok * E_);
      unsigned int* ap = (unsigned int*)(As + mrow * ASTR_F);
      for (int p = l16; p < E_ / 2; p += 16) ap[p] = ep[p];
    }
    __syncthreads();

    // stage 2: gates = [x_t | h] @ Wcat^T   (K = 576, 8 N-tiles per wave)
    v8f acc[8];
#pragma unroll
    for (int j = 0; j < 8; ++j) acc[j] = (v8f){0, 0, 0, 0, 0, 0, 0, 0};
    for (int kb = 0; kb < KB_TOKF; ++kb) {
      const unsigned short* arow = As + n * ASTR_F + kb * 32 + 8 * half;
      u32x4 a0 = *(const u32x4*)arow;
      u32x4 a1 = *(const u32x4*)(arow + 16);
      v16bf af = fr2(a0, a1);
#pragma unroll
      for (int j = 0; j < 8; ++j) {
        const u32x8* bp = (const u32x8*)(
            Wt + (((w + 8 * j) * KB_TOKF + kb) * 512 + lane * 16));
        acc[j] = wmma_bf16(af, fr1(*bp), acc[j]);
      }
    }
    __syncthreads();

    // stage 3: LSTM cell in registers; masked update; h -> A[:,300+j] bf16
#pragma unroll
    for (int u = 0; u < 2; ++u) {
#pragma unroll
      for (int r = 0; r < 8; ++r) {
        float iv = acc[0 + u][r] + bj[0 + u];
        float fv = acc[2 + u][r] + bj[2 + u];
        float gv = acc[4 + u][r] + bj[4 + u];
        float ov = acc[6 + u][r] + bj[6 + u];
        float cn = sigm(fv) * cst[u][r] + sigm(iv) * tanhf(gv);
        float hn = sigm(ov) * tanhf(cn);
        if (t < clen8[r]) {
          cst[u][r] = cn;
          int m = r + 8 * half;
          As[m * ASTR_F + 300 + u * 128 + 16 * w + n] = f2bf(hn);
        }
      }
    }
    __syncthreads();
  }
  // emit final h (frozen at clen-1) -> chunks[:, 0:256)
  for (int i = tid; i < 16 * HT_; i += 256) {
    int m = i >> 8, j = i & 255;
    chunks[(n0 + m) * 512 + j] = As[m * ASTR_F + 300 + j];
  }
}

// ---------------------------------------------------------------------------
// Token-level backward: hb[clen-1] == ONE step on x[clen-1] from zero state.
// ---------------------------------------------------------------------------
__global__ __launch_bounds__(256) void k_token_bwd(
    const int* __restrict__ x, const int* __restrict__ xclen,
    const unsigned short* __restrict__ embbf,
    const unsigned short* __restrict__ Wt,   // swizzled [NB][KB_TOKB][512]
    const float* __restrict__ bias,
    unsigned short* __restrict__ chunks) {   // cols 256..511
  __shared__ unsigned short As[16 * ASTR_B];
  const int tid = threadIdx.x;
  const int n0 = blockIdx.x * 16;
  const int w = tid >> 5, lane = tid & 31;
  const int n = lane & 15, half = lane >> 4;

  for (int i = tid; i < 16 * ASTR_B; i += 256) As[i] = 0;

  float bj[8];
#pragma unroll
  for (int j = 0; j < 8; ++j) bj[j] = bias[16 * (w + 8 * j) + n];

  const int mrow = tid >> 4;
  const int l16 = tid & 15;
  {
    int cl = clampT(xclen[n0 + mrow]);
    int tok = x[(n0 + mrow) * T_ + cl - 1];
    const unsigned int* ep = (const unsigned int*)(embbf + tok * E_);
    unsigned int* ap = (unsigned int*)(As + mrow * ASTR_B);
    for (int p = l16; p < E_ / 2; p += 16) ap[p] = ep[p];
  }
  __syncthreads();

  v8f acc[8];
#pragma unroll
  for (int j = 0; j < 8; ++j) acc[j] = (v8f){0, 0, 0, 0, 0, 0, 0, 0};
  for (int kb = 0; kb < KB_TOKB; ++kb) {
    const unsigned short* arow = As + n * ASTR_B + kb * 32 + 8 * half;
    u32x4 a0 = *(const u32x4*)arow;
    u32x4 a1 = *(const u32x4*)(arow + 16);
    v16bf af = fr2(a0, a1);
#pragma unroll
    for (int j = 0; j < 8; ++j) {
      const u32x8* bp = (const u32x8*)(
          Wt + (((w + 8 * j) * KB_TOKB + kb) * 512 + lane * 16));
      acc[j] = wmma_bf16(af, fr1(*bp), acc[j]);
    }
  }

#pragma unroll
  for (int u = 0; u < 2; ++u) {
#pragma unroll
    for (int r = 0; r < 8; ++r) {
      float iv = acc[0 + u][r] + bj[0 + u];
      float fv = acc[2 + u][r] + bj[2 + u];
      float gv = acc[4 + u][r] + bj[4 + u];
      float ov = acc[6 + u][r] + bj[6 + u];
      (void)fv;  // c_old = 0 => forget path vanishes
      float cn = sigm(iv) * tanhf(gv);
      float hn = sigm(ov) * tanhf(cn);
      int m = r + 8 * half;
      chunks[(n0 + m) * 512 + 256 + u * 128 + 16 * w + n] = f2bf(hn);
    }
  }
}

// ---------------------------------------------------------------------------
// Chunk-level BiLSTM: 32 batch seqs, 32 steps, outputs all steps (0 at pads).
// grid 4: blockIdx bit0 = row tile (16 seqs), bit1 = direction.
// ---------------------------------------------------------------------------
__global__ __launch_bounds__(256) void k_chunk(
    const unsigned short* __restrict__ chunks,  // [1024][512] bf16
    const int* __restrict__ xlen,               // [32]
    const unsigned short* __restrict__ Wf, const unsigned short* __restrict__ Wb,
    const float* __restrict__ bfwd, const float* __restrict__ bbwd,
    float* __restrict__ ch) {                   // [32][32][512] f32
  __shared__ unsigned short As[16 * ASTR_C];
  const int tid = threadIdx.x;
  const int dir = blockIdx.x >> 1;
  const int n0 = (blockIdx.x & 1) * 16;
  const unsigned short* Wt = dir ? Wb : Wf;
  const float* bias = dir ? bbwd : bfwd;
  const int w = tid >> 5, lane = tid & 31;
  const int n = lane & 15, half = lane >> 4;

  for (int i = tid; i < 16 * ASTR_C; i += 256) As[i] = 0;

  float bj[8];
#pragma unroll
  for (int j = 0; j < 8; ++j) bj[j] = bias[16 * (w + 8 * j) + n];
  int len8[8];
#pragma unroll
  for (int r = 0; r < 8; ++r) len8[r] = xlen[n0 + r + 8 * half];
  float cst[2][8];
#pragma unroll
  for (int u = 0; u < 2; ++u)
#pragma unroll
    for (int r = 0; r < 8; ++r) cst[u][r] = 0.f;

  const int mrow = tid >> 4;
  const int l16 = tid & 15;

  for (int step = 0; step < LC_; ++step) {
    const int l = dir ? (LC_ - 1 - step) : step;
    {  // stage 1: chunk features -> A[:,0:512)
      const unsigned int* sp =
          (const unsigned int*)(chunks + ((n0 + mrow) * LC_ + l) * 512);
      unsigned int* ap = (unsigned int*)(As + mrow * ASTR_C);
      for (int p = l16; p < 256; p += 16) ap[p] = sp[p];
    }
    __syncthreads();

    v8f acc[8];
#pragma unroll
    for (int j = 0; j < 8; ++j) acc[j] = (v8f){0, 0, 0, 0, 0, 0, 0, 0};
    for (int kb = 0; kb < KB_CHK; ++kb) {
      const unsigned short* arow = As + n * ASTR_C + kb * 32 + 8 * half;
      u32x4 a0 = *(const u32x4*)arow;
      u32x4 a1 = *(const u32x4*)(arow + 16);
      v16bf af = fr2(a0, a1);
#pragma unroll
      for (int j = 0; j < 8; ++j) {
        const u32x8* bp = (const u32x8*)(
            Wt + (((w + 8 * j) * KB_CHK + kb) * 512 + lane * 16));
        acc[j] = wmma_bf16(af, fr1(*bp), acc[j]);
      }
    }
    __syncthreads();

#pragma unroll
    for (int u = 0; u < 2; ++u) {
#pragma unroll
      for (int r = 0; r < 8; ++r) {
        float iv = acc[0 + u][r] + bj[0 + u];
        float fv = acc[2 + u][r] + bj[2 + u];
        float gv = acc[4 + u][r] + bj[4 + u];
        float ov = acc[6 + u][r] + bj[6 + u];
        float cn = sigm(fv) * cst[u][r] + sigm(iv) * tanhf(gv);
        float hn = sigm(ov) * tanhf(cn);
        int m = r + 8 * half;
        int junit = u * 128 + 16 * w + n;
        bool valid = l < len8[r];
        if (valid) {
          cst[u][r] = cn;
          As[m * ASTR_C + 512 + junit] = f2bf(hn);
        }
        ch[((n0 + m) * LC_ + l) * 512 + dir * 256 + junit] = valid ? hn : 0.0f;
      }
    }
    __syncthreads();
  }
}

// ---------------------------------------------------------------------------
// FC: out[b,l,k] = ch[b,l,:] . fc_W[k,:] + fc_b[k]
// ---------------------------------------------------------------------------
__global__ void k_fc(const float* __restrict__ ch, const float* __restrict__ W,
                     const float* __restrict__ b, float* __restrict__ out) {
  int i = blockIdx.x * blockDim.x + threadIdx.x;
  if (i >= B_ * LC_ * NTAG_) return;
  int k = i % NTAG_, bl = i / NTAG_;
  const float* v = ch + bl * 512;
  const float* wr = W + k * 512;
  float s = b[k];
  for (int c = 0; c < 512; ++c) s += v[c] * wr[c];
  out[i] = s;
}

// ---------------------------------------------------------------------------
extern "C" void kernel_launch(void* const* d_in, const int* in_sizes, int n_in,
                              void* d_out, int out_size, void* d_ws,
                              size_t ws_size, hipStream_t stream) {
  const int*   x          = (const int*)d_in[0];
  // d_in[1] = x_feats : unused by the reference
  const int*   x_len      = (const int*)d_in[2];
  const int*   x_clen     = (const int*)d_in[3];
  const float* emb        = (const float*)d_in[4];
  const float* tok_Wih_f  = (const float*)d_in[5];
  const float* tok_Whh_f  = (const float*)d_in[6];
  const float* tok_b_f    = (const float*)d_in[7];
  const float* tok_Wih_b  = (const float*)d_in[8];
  // d_in[9] tok_Whh_b, d_in[10] tok_b_b
  const float* tok_b_b    = (const float*)d_in[10];
  const float* chk_Wih_f  = (const float*)d_in[11];
  const float* chk_Whh_f  = (const float*)d_in[12];
  const float* chk_b_f    = (const float*)d_in[13];
  const float* chk_Wih_b  = (const float*)d_in[14];
  const float* chk_Whh_b  = (const float*)d_in[15];
  const float* chk_b_b    = (const float*)d_in[16];
  const float* fc_W       = (const float*)d_in[17];
  const float* fc_b       = (const float*)d_in[18];

  unsigned char* ws = (unsigned char*)d_ws;
  size_t off = 0;
  auto take = [&](size_t bytes) {
    unsigned char* p = ws + off;
    off = (off + bytes + 255) & ~(size_t)255;
    return p;
  };
  unsigned short* embbf  = (unsigned short*)take((size_t)V_ * E_ * 2);        // 18.0 MB
  unsigned short* tokWf  = (unsigned short*)take((size_t)NB * KB_TOKF * 1024);// 1.18 MB
  unsigned short* tokWb  = (unsigned short*)take((size_t)NB * KB_TOKB * 1024);// 0.66 MB
  unsigned short* chkWf  = (unsigned short*)take((size_t)NB * KB_CHK * 1024); // 1.57 MB
  unsigned short* chkWb  = (unsigned short*)take((size_t)NB * KB_CHK * 1024); // 1.57 MB
  unsigned short* chunks = (unsigned short*)take((size_t)NSEQ * 512 * 2);     // 1.05 MB
  float*          chbuf  = (float*)take((size_t)NSEQ * 512 * 4);              // 2.10 MB
  (void)ws_size; (void)in_sizes; (void)n_in; (void)out_size;

  k_f32_to_bf16<<<2048, 256, 0, stream>>>(emb, embbf, V_ * E_);
  k_swizzle_w<<<(NB * KB_TOKF * 512 + 255) / 256, 256, 0, stream>>>(
      tok_Wih_f, tok_Whh_f, 300, 256, KB_TOKF, tokWf);
  k_swizzle_w<<<(NB * KB_TOKB * 512 + 255) / 256, 256, 0, stream>>>(
      tok_Wih_b, nullptr, 300, 0, KB_TOKB, tokWb);
  k_swizzle_w<<<(NB * KB_CHK * 512 + 255) / 256, 256, 0, stream>>>(
      chk_Wih_f, chk_Whh_f, 512, 256, KB_CHK, chkWf);
  k_swizzle_w<<<(NB * KB_CHK * 512 + 255) / 256, 256, 0, stream>>>(
      chk_Wih_b, chk_Whh_b, 512, 256, KB_CHK, chkWb);

  k_token_fwd<<<NSEQ / 16, 256, 0, stream>>>(x, x_clen, embbf, tokWf, tok_b_f,
                                             chunks);
  k_token_bwd<<<NSEQ / 16, 256, 0, stream>>>(x, x_clen, embbf, tokWb, tok_b_b,
                                             chunks);
  k_chunk<<<4, 256, 0, stream>>>(chunks, x_len, chkWf, chkWb, chk_b_f, chk_b_b,
                                 chbuf);
  k_fc<<<(B_ * LC_ * NTAG_ + 255) / 256, 256, 0, stream>>>(chbuf, fc_W, fc_b,
                                                           (float*)d_out);
}
